// HEA_10462540333708
// MI455X (gfx1250) — matrix operand order repaired and checked by reading, not compiled, not run on hardware
//
#include <hip/hip_runtime.h>

// Problem constants (match reference)
#define B_   16384
#define T_   2
#define D_   1024
#define DE_  512
#define S_   4
#define P_   4
#define NE   8      // S_ + P_ experts per task
#define NM   12     // unique expert matrices: 4 shared + 2*4 specific

// Tiling for the fused MoE GEMM
#define MT   256    // rows per block (4 waves x 64)
#define NT   128    // DE cols per block (2 waves x 64)
#define KT   64     // K per LDS stage (2 WMMA K-chunks)

typedef __bf16 bf16_t;
typedef bf16_t v8bf  __attribute__((ext_vector_type(8)));
typedef bf16_t v16bf __attribute__((ext_vector_type(16)));
typedef float  v8f   __attribute__((ext_vector_type(8)));

// workspace layout (bytes)
#define GW_BYTES  ((size_t)B_ * T_ * NE * 4)            //  1,048,576
#define XB_BYTES  ((size_t)T_ * B_ * D_ * 2)            // 67,108,864
// wtb starts at GW_BYTES + XB_BYTES (64B aligned)

// Truncate a flat pointer to the wave-relative LDS byte offset
// (ISA 10.2: LDS aperture flat addresses carry the LDS offset in [31:0])
__device__ __forceinline__ unsigned lds_off(const void* p) {
  return (unsigned)(size_t)p;
}

// -------------------------------------------------------------------------
// Pre-pass A: x (f32) -> xb (bf16) [T][B][D]
// -------------------------------------------------------------------------
__global__ __launch_bounds__(256) void hea_cvt_x(
    const float* __restrict__ x0, const float* __restrict__ x1,
    bf16_t* __restrict__ xb) {
  const int t = blockIdx.y;
  size_t off = ((size_t)blockIdx.x * 256 + threadIdx.x) * 8;
  const float* src = (t ? x1 : x0) + off;
  float4 a = *(const float4*)src;
  float4 b = *(const float4*)(src + 4);
  v8bf o;
  o[0] = (bf16_t)a.x; o[1] = (bf16_t)a.y; o[2] = (bf16_t)a.z; o[3] = (bf16_t)a.w;
  o[4] = (bf16_t)b.x; o[5] = (bf16_t)b.y; o[6] = (bf16_t)b.z; o[7] = (bf16_t)b.w;
  *(v8bf*)(xb + (size_t)t * B_ * D_ + off) = o;
}

// -------------------------------------------------------------------------
// Pre-pass B: W_share / W_spcf (f32, [D][DE]) -> wtb (bf16, [NM][DE][D])
// matrix m: 0..3 shared, 4..11 = spcf flat (t*P + p)
// -------------------------------------------------------------------------
__global__ __launch_bounds__(256) void hea_cvt_w(
    const float* __restrict__ Wsh, const float* __restrict__ Wsp,
    bf16_t* __restrict__ wtb) {
  size_t g = (size_t)blockIdx.x * 256 + threadIdx.x;   // NM*DE*(D/8) threads
  int f = (int)(g % DE_);
  size_t rest = g / DE_;
  int d0 = (int)(rest % (D_ / 8)) * 8;
  int m  = (int)(rest / (D_ / 8));
  const float* src = (m < S_) ? (Wsh + (size_t)m * D_ * DE_)
                              : (Wsp + (size_t)(m - S_) * D_ * DE_);
  v8bf o;
#pragma unroll
  for (int i = 0; i < 8; ++i)
    o[i] = (bf16_t)src[(size_t)(d0 + i) * DE_ + f];    // coalesced across f
  *(v8bf*)(wtb + ((size_t)m * DE_ + f) * D_ + d0) = o;
}

// -------------------------------------------------------------------------
// Kernel 1: gates[b,t,:] = softmax(x[b,t,:] @ W_gate[t] + b_gate[t])
// one wave32 per (b,t) row
// -------------------------------------------------------------------------
__global__ __launch_bounds__(256) void hea_gates(
    const float* __restrict__ x0, const float* __restrict__ x1,
    const float* __restrict__ Wg, const float* __restrict__ bg,
    float* __restrict__ gw) {
  const int wid  = threadIdx.x >> 5;
  const int lane = threadIdx.x & 31;
  const int rid  = blockIdx.x * 8 + wid;      // row id in [0, B*T)
  const int b = rid >> 1, t = rid & 1;
  const float* x = (t ? x1 : x0) + (size_t)b * D_;

  float acc[NE];
#pragma unroll
  for (int e = 0; e < NE; ++e) acc[e] = 0.f;

  for (int d = lane; d < D_; d += 32) {
    float xv = x[d];
    const float4* w = (const float4*)(Wg + ((size_t)t * D_ + d) * NE);
    float4 w0 = w[0], w1 = w[1];
    acc[0] += xv * w0.x; acc[1] += xv * w0.y;
    acc[2] += xv * w0.z; acc[3] += xv * w0.w;
    acc[4] += xv * w1.x; acc[5] += xv * w1.y;
    acc[6] += xv * w1.z; acc[7] += xv * w1.w;
  }
#pragma unroll
  for (int off = 16; off >= 1; off >>= 1) {
#pragma unroll
    for (int e = 0; e < NE; ++e) acc[e] += __shfl_xor(acc[e], off, 32);
  }
#pragma unroll
  for (int e = 0; e < NE; ++e) acc[e] += bg[t * NE + e];

  float m = acc[0];
#pragma unroll
  for (int e = 1; e < NE; ++e) m = fmaxf(m, acc[e]);
  float ex[NE], s = 0.f;
#pragma unroll
  for (int e = 0; e < NE; ++e) { ex[e] = __expf(acc[e] - m); s += ex[e]; }
  float inv = 1.f / s;
  if (lane < NE) gw[(size_t)rid * NE + lane] = ex[lane] * inv;
}

// -------------------------------------------------------------------------
// Kernel 2: fused 8-expert GEMM + bias + relu + softmax-gated mixture
// block = 256 threads = 8 waves (4x2), tile 256x128, K staged 64 at a time.
// Each wave owns a 64x64 macro-tile (4x4 WMMA tiles -> 4x fragment reuse).
// X tile staged via global_load_async_to_lds_b128 (ASYNCcnt);
// B fragments read directly from pre-transposed bf16 weights (L2-resident).
// -------------------------------------------------------------------------
__global__ __launch_bounds__(256) void hea_moe(
    const bf16_t* __restrict__ xb, const bf16_t* __restrict__ wtb,
    const float* __restrict__ bsh, const float* __restrict__ bsp,
    const float* __restrict__ gw,
    float* __restrict__ out) {
  __shared__ __align__(32) bf16_t sX[MT * KT];   // X tile [row][k]   (32 KB)
  __shared__ __align__(16) float  sG[MT * NE];   // gates  [row][e]   ( 8 KB)

  const int tid = threadIdx.x;
  const int t   = blockIdx.z;
  const int fb  = blockIdx.y * NT;
  const int bb  = blockIdx.x * MT;
  const bf16_t* X = xb + (size_t)t * B_ * D_;

  // stage gates for this row tile (2048 floats, 8 per thread)
  {
#pragma unroll
    for (int jj = 0; jj < 2; ++jj) {
      int gidx = (jj * 256 + tid) * 4;
      int i = gidx >> 3, e0 = gidx & 7;
      float4 g = *(const float4*)(gw + ((size_t)(bb + i) * T_ + t) * NE + e0);
      *(float4*)&sG[gidx] = g;
    }
  }

  const int  wave = tid >> 5;
  const int  lane = tid & 31;
  const int  wm   = wave >> 1;     // 0..3 : 64 rows each
  const int  wn   = wave & 1;      // 0..1 : 64 cols each
  const int  lrow = lane & 15;
  const bool lo16 = (lane < 16);

  v8f oacc[4][4];
#pragma unroll
  for (int mt = 0; mt < 4; ++mt)
#pragma unroll
    for (int nt = 0; nt < 4; ++nt)
#pragma unroll
      for (int r = 0; r < 8; ++r) oacc[mt][nt][r] = 0.f;

  for (int e = 0; e < NE; ++e) {
    const int m = (e < S_) ? e : (S_ + t * P_ + (e - S_));
    const bf16_t* Wt   = wtb + (size_t)m * DE_ * D_;       // [DE][D] bf16
    const float*  bias = (e < S_) ? (bsh + e * DE_)
                                  : (bsp + (t * P_ + (e - S_)) * DE_);

    v8f acc[4][4];
#pragma unroll
    for (int mt = 0; mt < 4; ++mt)
#pragma unroll
      for (int nt = 0; nt < 4; ++nt)
#pragma unroll
        for (int r = 0; r < 8; ++r) acc[mt][nt][r] = 0.f;

    for (int kb = 0; kb < D_; kb += KT) {
      __syncthreads();   // previous stage's LDS reads done before overwrite

      // ---- async-stage X tile: 256x64 bf16 (8 x 16B per thread) ----
#pragma unroll
      for (int j = 0; j < 8; ++j) {
        int c  = j * 256 + tid;            // 2048 chunks of 8 bf16
        int r  = c >> 3;
        int kc = (c & 7) * 8;
        const bf16_t* gp = X + (size_t)(bb + r) * D_ + kb + kc;
        unsigned loff = lds_off(&sX[r * KT + kc]);
        asm volatile("global_load_async_to_lds_b128 %0, %1, off"
                     :
                     : "v"(loff), "v"(gp)
                     : "memory");
      }
      asm volatile("s_wait_asynccnt 0x0" ::: "memory");
      __syncthreads();

#pragma unroll
      for (int kc2 = 0; kc2 < KT; kc2 += 32) {
        // A fragments from LDS (contiguous k = 16*(lane/16) + s, same order as B)
        v16bf af[4];
#pragma unroll
        for (int mt = 0; mt < 4; ++mt) {
          int ar = wm * 64 + mt * 16 + lrow;
          int k0 = kc2 + (lo16 ? 0 : 16);
          v8bf alo = *(const v8bf*)&sX[ar * KT + k0];
          v8bf ahi = *(const v8bf*)&sX[ar * KT + k0 + 8];
          af[mt] = __builtin_shufflevector(alo, ahi,
                   0, 1, 2, 3, 4, 5, 6, 7, 8, 9, 10, 11, 12, 13, 14, 15);
        }
        // B fragments straight from pre-transposed bf16 weights
        v16bf bq[4];
#pragma unroll
        for (int nt = 0; nt < 4; ++nt) {
          int n  = fb + wn * 64 + nt * 16 + lrow;
          int gk = kb + kc2 + (lo16 ? 0 : 16);
          bq[nt] = *(const v16bf*)(Wt + (size_t)n * D_ + gk);
        }
#pragma unroll
        for (int mt = 0; mt < 4; ++mt)
#pragma unroll
          for (int nt = 0; nt < 4; ++nt)
            acc[mt][nt] = __builtin_amdgcn_wmma_f32_16x16x32_bf16(
                false, af[mt], false, bq[nt], (short)0, acc[mt][nt],
                false, false);
      }
    }

    // ---- fold expert e: out += gate[row,e] * relu(acc + bias) ----
#pragma unroll
    for (int mt = 0; mt < 4; ++mt) {
#pragma unroll
      for (int nt = 0; nt < 4; ++nt) {
        int ncol = wn * 64 + nt * 16 + lrow;
        float bv = bias[fb + ncol];
#pragma unroll
        for (int r = 0; r < 8; ++r) {
          int rt = wm * 64 + mt * 16 + r + (lo16 ? 0 : 8);  // C: M=r+8*(lane>=16)
          float g = sG[rt * NE + e];
          float v = fmaxf(acc[mt][nt][r] + bv, 0.f);
          oacc[mt][nt][r] += g * v;
        }
      }
    }
  }

  // ---- store output [B, T, DE] ----
#pragma unroll
  for (int mt = 0; mt < 4; ++mt) {
#pragma unroll
    for (int nt = 0; nt < 4; ++nt) {
      int ncol = wn * 64 + nt * 16 + lrow;
#pragma unroll
      for (int r = 0; r < 8; ++r) {
        int row = bb + wm * 64 + mt * 16 + r + (lo16 ? 0 : 8);
        out[((size_t)row * T_ + t) * DE_ + fb + ncol] = oacc[mt][nt][r];
      }
    }
  }
}

// -------------------------------------------------------------------------
extern "C" void kernel_launch(void* const* d_in, const int* in_sizes, int n_in,
                              void* d_out, int out_size, void* d_ws, size_t ws_size,
                              hipStream_t stream) {
  const float* x0  = (const float*)d_in[0];
  const float* x1  = (const float*)d_in[1];
  const float* Wsh = (const float*)d_in[2];
  const float* bsh = (const float*)d_in[3];
  const float* Wsp = (const float*)d_in[4];
  const float* bsp = (const float*)d_in[5];
  const float* Wg  = (const float*)d_in[6];
  const float* bg  = (const float*)d_in[7];
  float* out = (float*)d_out;

  float*  gw  = (float*)d_ws;                                   // gates
  bf16_t* xb  = (bf16_t*)((char*)d_ws + GW_BYTES);              // bf16 x
  bf16_t* wtb = (bf16_t*)((char*)d_ws + GW_BYTES + XB_BYTES);   // bf16 W^T

  hea_cvt_x<<<dim3(B_ * D_ / 2048, T_), dim3(256), 0, stream>>>(x0, x1, xb);
  hea_cvt_w<<<dim3(NM * DE_ * (D_ / 8) / 256), dim3(256), 0, stream>>>(Wsh, Wsp, wtb);
  hea_gates<<<dim3((B_ * T_) / 8), dim3(256), 0, stream>>>(x0, x1, Wg, bg, gw);
  hea_moe<<<dim3(B_ / MT, DE_ / NT, T_), dim3(256), 0, stream>>>(
      xb, wtb, bsh, bsp, gw, out);
}
